// XLSR_AASIST_5068061409674
// MI455X (gfx1250) — matrix operand order repaired and checked by reading, not compiled, not run on hardware
//
#include <hip/hip_runtime.h>
#include <hip/hip_bf16.h>
#include <math.h>
#include <stdint.h>

typedef __attribute__((ext_vector_type(16))) _Float16 v16h;
typedef __attribute__((ext_vector_type(8)))  _Float16 v8h;
typedef __attribute__((ext_vector_type(8)))  float    v8f;
typedef __attribute__((ext_vector_type(4)))  float    v4f;
typedef __attribute__((ext_vector_type(4)))  uint32_t u32x4;
typedef __attribute__((ext_vector_type(8)))  uint32_t u32x8;

#define Bq 16
#define Nq 256
#define Dq 128

// Native gfx1250 transcendentals (v_tanh_f32 is a TRANS32 op on CDNA5 and
// co-executes with WMMA; avoids the branchy libm expansion w/ EXEC diamonds).
#if __has_builtin(__builtin_amdgcn_tanhf)
static __device__ inline float fast_tanh(float x) { return __builtin_amdgcn_tanhf(x); }
#else
static __device__ inline float fast_tanh(float x) { return tanhf(x); }
#endif
static __device__ inline float fast_exp(float x) { return __expf(x); }

// ---------------------------------------------------------------- WMMA core
static __device__ inline v8f wmma_f16(v16h a, v16h b, v8f c) {
  return __builtin_amdgcn_wmma_f32_16x16x32_f16(
      /*neg_a=*/false, a, /*neg_b=*/false, b,
      /*c_mod=*/(short)0, c, /*reuse_a=*/false, /*reuse_b=*/false);
}

// A-fragment (16x32 f16): lane holds row M=lane&15.
// element e -> K = kc*32 + (e>>3)*16 + 8*(lane>>4) + (e&7)
static __device__ inline v16h afrag_h(const _Float16* rowp, int off /* kc*32 + hi8 */) {
  v8h h0 = *(const v8h*)(rowp + off);
  v8h h1 = *(const v8h*)(rowp + off + 16);
  v16h r;
#pragma unroll
  for (int e = 0; e < 8; ++e) { r[e] = h0[e]; r[e + 8] = h1[e]; }
  return r;
}

// B-fragment (32x16 f16): lane holds column N=lane&15.
// element e -> K = kc*32 + (e>>3)*16 + 8*(lane>>4) + (e&7)
static __device__ inline v16h bfrag_h(const _Float16* rowp, int off /* kc*32 + hi8 */) {
  v8h h0 = *(const v8h*)(rowp + off);
  v8h h1 = *(const v8h*)(rowp + off + 16);
  v16h r;
#pragma unroll
  for (int e = 0; e < 8; ++e) { r[e] = h0[e]; r[e + 8] = h1[e]; }
  return r;
}

// ------------------------------------------------- CDNA5 async global->LDS
static __device__ inline void async_b128(uint32_t lds_byte_off, const void* gaddr) {
  asm volatile("global_load_async_to_lds_b128 %0, %1, off"
               :: "v"(lds_byte_off), "v"(gaddr) : "memory");
}
static __device__ inline void wait_async0() {
  asm volatile("s_wait_asynccnt 0x0" ::: "memory");
}

// ============================================================== Kernel 0
// One-time f16 conversion + pre-transposition so all later LDS staging is a
// contiguous byte copy (TDM / async-load friendly) and the hot kernels carry
// no f32->f16 conversions.
__global__ __launch_bounds__(256)
void k_prep(const float* __restrict__ x, const float* __restrict__ apw,
            const float* __restrict__ pwa_w, const float* __restrict__ pwoa_w,
            _Float16* __restrict__ xh, _Float16* __restrict__ xhT,
            _Float16* __restrict__ wth, _Float16* __restrict__ pwaT,
            _Float16* __restrict__ pwoaT) {
  const int idx = blockIdx.x * 256 + threadIdx.x;
  const int total_x = Bq * Nq * Dq;                   // 524288
  if (idx < total_x) {
    const float v = x[idx];
    const int d = idx & 127;
    const int j = (idx >> 7) & 255;
    const int b = idx >> 15;
    xh[idx] = (_Float16)v;                            // xh[b][j][d]
    xhT[((size_t)b * Dq + d) * Nq + j] = (_Float16)v; // xhT[b][d][j]
  } else {
    const int w = idx - total_x;                      // 0..49151
    if (w < 3 * Dq * Dq) {
      const int which = w >> 14;
      const int e = w & 16383;
      const int n = e >> 7, k = e & 127;
      const float* src = (which == 0) ? apw : ((which == 1) ? pwa_w : pwoa_w);
      _Float16* dst    = (which == 0) ? wth : ((which == 1) ? pwaT : pwoaT);
      dst[e] = (_Float16)src[k * Dq + n];             // dst[n][k] = src[k][n]
    }
  }
}

// ============================================================== Kernel 1
// Bilinear attention logits + column softmax.
// Grid: Bq*(Nq/16) WGs of 256 threads. WG (b, j-tile of 16).
// Wt staged by the Tensor Data Mover (32 KB contiguous f16 tile).
__global__ __launch_bounds__(256)
void k_attn(const _Float16* __restrict__ xh, const _Float16* __restrict__ wth,
            const float* __restrict__ apb, const float* __restrict__ aw,
            _Float16* __restrict__ att) {
  __shared__ _Float16 Wt[Dq * Dq];   // 32 KB : Wt[n][k]
  __shared__ float col[Nq];
  __shared__ float ecol[Nq];

  const int tid = threadIdx.x;
  const int b   = blockIdx.x >> 4;
  const int j0  = (blockIdx.x & 15) * 16;
  const int lane = tid & 31;
  const int wave = tid >> 5;
  const int nl   = lane & 15;
  const int hi8  = (lane >> 4) << 3;
  const int hi16 = hi8 << 1;

  // ---- TDM: DMA the 32KB Wt tile from global to LDS (wave 0 issues) ----
  if (wave == 0) {
    uint32_t ldsa = (uint32_t)__builtin_amdgcn_readfirstlane(
        (uint32_t)(uintptr_t)&Wt[0]);
    uint64_t ga   = (uint64_t)(uintptr_t)wth;
    uint32_t galo = (uint32_t)__builtin_amdgcn_readfirstlane((uint32_t)ga);
    uint32_t gahi = (uint32_t)__builtin_amdgcn_readfirstlane((uint32_t)(ga >> 32));
    u32x4 g0;
    g0[0] = 1u;                                       // count=1, user mode
    g0[1] = ldsa;                                     // lds_addr
    g0[2] = galo;                                     // global_addr[31:0]
    g0[3] = (gahi & 0x01FFFFFFu) | 0x80000000u;       // addr[56:32] | type=2
    u32x8 g1;
    g1[0] = 0x00010000u;   // workgroup_mask=0, data_size=1 (2 bytes)
    g1[1] = 0x40000000u;   // tensor_dim0 = 16384 (bits 79:48, low half)
    g1[2] = 0x00010000u;   // tensor_dim0 hi = 0, tensor_dim1 = 1
    g1[3] = 0x40000000u;   // tile_dim0 = 16384 (bits 127:112)
    g1[4] = 0x00000001u;   // tile_dim1 = 1, tile_dim2 = 0
    g1[5] = 16384u;        // tensor_dim0_stride (bits 207:160, low word)
    g1[6] = 0u;
    g1[7] = 0u;
    asm volatile("tensor_load_to_lds %0, %1" :: "s"(g0), "s"(g1) : "memory");
    __builtin_amdgcn_s_wait_tensorcnt(0);
  }
  __syncthreads();

  // Persistent A fragments: this wave's two 16-row i-tiles of x_b (f16)
  const _Float16* xb = xh + (size_t)b * Nq * Dq;
  v16h afr[2][4];
#pragma unroll
  for (int t = 0; t < 2; ++t) {
    const _Float16* rowp = xb + ((wave * 2 + t) * 16 + nl) * Dq;
#pragma unroll
    for (int kc = 0; kc < 4; ++kc)
      afr[t][kc] = afrag_h(rowp, kc * 32 + hi8);
  }
  // j-invariant epilogue constants
  float biasv[8], awnv[8];
#pragma unroll
  for (int nt = 0; nt < 8; ++nt) {
    biasv[nt] = apb[nt * 16 + nl];
    awnv[nt]  = aw[nt * 16 + nl];
  }

  for (int jj = 0; jj < 16; ++jj) {
    const int j = j0 + jj;
    const _Float16* xj = xb + j * Dq;
    v16h xjf[4];
#pragma unroll
    for (int kc = 0; kc < 4; ++kc)
      xjf[kc] = bfrag_h(xj, kc * 32 + hi8);

    float acc0[8] = {0.f,0.f,0.f,0.f,0.f,0.f,0.f,0.f};
    float acc1[8] = {0.f,0.f,0.f,0.f,0.f,0.f,0.f,0.f};
#pragma unroll
    for (int nt = 0; nt < 8; ++nt) {
      v8f c0 = {}, c1 = {};
      const _Float16* wrow = Wt + (nt * 16 + nl) * Dq;
#pragma unroll
      for (int kc = 0; kc < 4; ++kc) {
        v16h bf = bfrag_h(wrow, kc * 32 + hi8);
        bf = bf * xjf[kc];               // scale B by x_j over K (packed f16)
        c0 = wmma_f16(afr[0][kc], bf, c0);   // one B fragment feeds
        c1 = wmma_f16(afr[1][kc], bf, c1);   // two i-tiles
      }
#pragma unroll
      for (int r = 0; r < 8; ++r) {
        acc0[r] += fast_tanh(c0[r] + biasv[nt]) * awnv[nt];  // v_tanh_f32
        acc1[r] += fast_tanh(c1[r] + biasv[nt]) * awnv[nt];
      }
    }
#pragma unroll
    for (int r = 0; r < 8; ++r) {
#pragma unroll
      for (int off = 1; off < 16; off <<= 1) {
        acc0[r] += __shfl_xor(acc0[r], off, 32);
        acc1[r] += __shfl_xor(acc1[r], off, 32);
      }
    }
    if (nl == 0) {
      const int i0a = wave * 32 + hi8;
#pragma unroll
      for (int r = 0; r < 8; ++r) {
        col[i0a + r]      = acc0[r];
        col[i0a + 16 + r] = acc1[r];
      }
    }
    __syncthreads();
    // softmax over the i-axis (full column owned by this WG)
    float m = -3.4e38f;
    for (int i = 0; i < Nq; ++i) m = fmaxf(m, col[i]);
    const float e = fast_exp(col[tid] - m);
    ecol[tid] = e;
    __syncthreads();
    float s = 0.f;
    for (int i = 0; i < Nq; ++i) s += ecol[i];
    att[((size_t)b * Nq + tid) * Nq + j] = (_Float16)(e / s);
    __syncthreads();
  }
}

// ============================================================== Kernel 2
// agg = att@x ; h = selu(BN(agg@pwa_w + x@pwoa_w + biases)) ; pool scores.
// All LDS staging via CDNA5 async global->LDS copies (ASYNCcnt path).
__global__ __launch_bounds__(256)
void k_proj(const _Float16* __restrict__ xh, const _Float16* __restrict__ xhT,
            const _Float16* __restrict__ att,
            const _Float16* __restrict__ pwaT, const _Float16* __restrict__ pwoaT,
            const float* __restrict__ pwa_b, const float* __restrict__ pwoa_b,
            const float* __restrict__ bn_g, const float* __restrict__ bn_b,
            const float* __restrict__ bn_m, const float* __restrict__ bn_v,
            const float* __restrict__ pool_w, const float* __restrict__ pool_b,
            float* __restrict__ hbuf, float* __restrict__ sbuf) {
  __shared__ _Float16 lds[32768];   // 64 KB, re-used across phases

  const int tid  = threadIdx.x;
  const int b    = blockIdx.x >> 1;
  const int half = blockIdx.x & 1;
  const int lane = tid & 31;
  const int wave = tid >> 5;
  const int nl   = lane & 15;
  const int hi8  = (lane >> 4) << 3;
  const int hi16 = hi8 << 1;
  const int i0   = half * 128 + wave * 16;
  const uint32_t lbase = (uint32_t)(uintptr_t)&lds[0];

  // ---- phase 1: async-copy xT[d][j] (64 KB contiguous) into LDS ----
  {
    const char* src = (const char*)(xhT + (size_t)b * Dq * Nq);
#pragma unroll
    for (int it = 0; it < 16; ++it) {
      const uint32_t off = (uint32_t)(tid * 16 + it * 4096);
      async_b128(lbase + off, src + off);
    }
    wait_async0();
  }
  __syncthreads();

  v8f c[8];
#pragma unroll
  for (int nt = 0; nt < 8; ++nt) c[nt] = (v8f){};

  const _Float16* atrow = att + ((size_t)b * Nq + i0 + nl) * Nq;
#pragma unroll
  for (int kc = 0; kc < 8; ++kc) {           // K = 256 (j)
    v16h af = afrag_h(atrow, kc * 32 + hi8);
#pragma unroll
    for (int nt = 0; nt < 8; ++nt) {
      v16h bf = bfrag_h(lds + (nt * 16 + nl) * 256, kc * 32 + hi8);
      c[nt] = wmma_f16(af, bf, c[nt]);
    }
  }
  __syncthreads();

  // ---- phase 2a: agg tiles -> lds[0:32KB), pwa_wT async -> lds[32KB:64KB) ----
  {
    _Float16* ag = lds + wave * 2048;
#pragma unroll
    for (int nt = 0; nt < 8; ++nt)
#pragma unroll
      for (int r = 0; r < 8; ++r)
        ag[(r + hi8) * Dq + nt * 16 + nl] = (_Float16)c[nt][r];
  }
  {
    const char* src = (const char*)pwaT;
#pragma unroll
    for (int it = 0; it < 8; ++it) {
      const uint32_t off = (uint32_t)(tid * 16 + it * 4096);
      async_b128(lbase + 32768u + off, src + off);
    }
    wait_async0();
  }
  __syncthreads();

  v8f d[8];
#pragma unroll
  for (int nt = 0; nt < 8; ++nt) d[nt] = (v8f){};

  const _Float16* agrow = lds + wave * 2048 + nl * Dq;
#pragma unroll
  for (int kc = 0; kc < 4; ++kc) {           // K = 128 (d)
    v16h af = afrag_h(agrow, kc * 32 + hi8);
#pragma unroll
    for (int nt = 0; nt < 8; ++nt) {
      v16h bf = bfrag_h(lds + 16384 + (nt * 16 + nl) * Dq, kc * 32 + hi8);
      d[nt] = wmma_f16(af, bf, d[nt]);
    }
  }
  __syncthreads();

  // ---- phase 2b: pwoa_wT async -> lds[0:32KB); accumulate x@pwoa into d ----
  {
    const char* src = (const char*)pwoaT;
#pragma unroll
    for (int it = 0; it < 8; ++it) {
      const uint32_t off = (uint32_t)(tid * 16 + it * 4096);
      async_b128(lbase + off, src + off);
    }
    wait_async0();
  }
  __syncthreads();

  const _Float16* xrow = xh + ((size_t)b * Nq + i0 + nl) * Dq;
#pragma unroll
  for (int kc = 0; kc < 4; ++kc) {
    v16h af = afrag_h(xrow, kc * 32 + hi8);
#pragma unroll
    for (int nt = 0; nt < 8; ++nt) {
      v16h bf = bfrag_h(lds + (nt * 16 + nl) * Dq, kc * 32 + hi8);
      d[nt] = wmma_f16(af, bf, d[nt]);
    }
  }

  // ---- epilogue: bias + BN(eval) + SELU + pool score, fused ----
  const float kSeluS = 1.0507009873554805f;
  const float kSeluA = 1.6732632423543772f;
  float pacc[8] = {0.f,0.f,0.f,0.f,0.f,0.f,0.f,0.f};
#pragma unroll
  for (int nt = 0; nt < 8; ++nt) {
    const int n = nt * 16 + nl;
    const float bias = pwa_b[n] + pwoa_b[n];
    const float sc   = bn_g[n] * rsqrtf(bn_v[n] + 1e-5f);
    const float sh   = bn_b[n] - bn_m[n] * sc;
    const float pw   = pool_w[n];
#pragma unroll
    for (int r = 0; r < 8; ++r) {
      float h = d[nt][r] + bias;
      h = h * sc + sh;
      h = (h > 0.f) ? kSeluS * h : kSeluS * kSeluA * (fast_exp(h) - 1.f);
      pacc[r] += h * pw;
      hbuf[((size_t)b * Nq + i0 + r + hi8) * Dq + n] = h;
    }
  }
#pragma unroll
  for (int r = 0; r < 8; ++r) {
#pragma unroll
    for (int off = 1; off < 16; off <<= 1)
      pacc[r] += __shfl_xor(pacc[r], off, 32);
  }
  const float pb = pool_b[0];
  if (nl == 0) {
#pragma unroll
    for (int r = 0; r < 8; ++r)
      sbuf[b * Nq + i0 + r + hi8] = 1.f / (1.f + fast_exp(-(pacc[r] + pb)));
  }
}

// ============================================================== Kernel 3
// Stable top-k (k=128) by rank counting + gather of h*score rows.
__global__ __launch_bounds__(256)
void k_pool(const float* __restrict__ hbuf, const float* __restrict__ sbuf,
            float* __restrict__ out) {
  __shared__ float s[Nq];
  const int b = blockIdx.x, t = threadIdx.x;
  const float st = sbuf[b * Nq + t];
  s[t] = st;
  __syncthreads();
  int rank = 0;
  for (int j = 0; j < Nq; ++j) {
    const float sj = s[j];
    rank += (sj > st) || (sj == st && j < t);   // matches lax.top_k stable order
  }
  if (rank < 128) {
    const float* hr = hbuf + ((size_t)b * Nq + t) * Dq;
    float* orow     = out + ((size_t)b * 128 + rank) * Dq;
    for (int dd = 0; dd < Dq; dd += 4) {
      v4f v = *(const v4f*)(hr + dd);
#pragma unroll
      for (int e = 0; e < 4; ++e) v[e] *= st;
      *(v4f*)(orow + dd) = v;
    }
  }
}

// ============================================================== launch
extern "C" void kernel_launch(void* const* d_in, const int* in_sizes, int n_in,
                              void* d_out, int out_size, void* d_ws, size_t ws_size,
                              hipStream_t stream) {
  (void)in_sizes; (void)n_in; (void)out_size; (void)ws_size;
  const float* x      = (const float*)d_in[0];
  const float* apw    = (const float*)d_in[1];
  const float* apb    = (const float*)d_in[2];
  const float* aw     = (const float*)d_in[3];
  const float* pwa_w  = (const float*)d_in[4];
  const float* pwa_b  = (const float*)d_in[5];
  const float* pwoa_w = (const float*)d_in[6];
  const float* pwoa_b = (const float*)d_in[7];
  const float* bn_g   = (const float*)d_in[8];
  const float* bn_b   = (const float*)d_in[9];
  const float* bn_m   = (const float*)d_in[10];
  const float* bn_v   = (const float*)d_in[11];
  const float* pool_w = (const float*)d_in[12];
  const float* pool_b = (const float*)d_in[13];

  // ws layout:
  //  att   f16  [B][N][N]   2 MB
  //  hbuf  f32  [B][N][D]   2 MB
  //  sbuf  f32  [B][N]      16 KB
  //  xh    f16  [B][N][D]   1 MB
  //  xhT   f16  [B][D][N]   1 MB
  //  wth / pwaT / pwoaT f16 [D][D]  3 x 32 KB
  char* w = (char*)d_ws;
  _Float16* att  = (_Float16*)w;                       w += (size_t)Bq*Nq*Nq*2;
  float*    hbuf = (float*)w;                          w += (size_t)Bq*Nq*Dq*4;
  float*    sbuf = (float*)w;                          w += (size_t)Bq*Nq*4;
  _Float16* xh   = (_Float16*)w;                       w += (size_t)Bq*Nq*Dq*2;
  _Float16* xhT  = (_Float16*)w;                       w += (size_t)Bq*Nq*Dq*2;
  _Float16* wth  = (_Float16*)w;                       w += (size_t)Dq*Dq*2;
  _Float16* pwaT = (_Float16*)w;                       w += (size_t)Dq*Dq*2;
  _Float16* pwoaT= (_Float16*)w;

  const int prep_blocks = (Bq*Nq*Dq + 3*Dq*Dq + 255) / 256;
  k_prep<<<prep_blocks, 256, 0, stream>>>(x, apw, pwa_w, pwoa_w,
                                          xh, xhT, wth, pwaT, pwoaT);
  k_attn<<<Bq * (Nq / 16), 256, 0, stream>>>(xh, wth, apb, aw, att);
  k_proj<<<Bq * 2, 256, 0, stream>>>(xh, xhT, att, pwaT, pwoaT,
                                     pwa_b, pwoa_b, bn_g, bn_b, bn_m, bn_v,
                                     pool_w, pool_b, hbuf, sbuf);
  k_pool<<<Bq, 256, 0, stream>>>(hbuf, sbuf, (float*)d_out);
}
